// FractalModel_919123001781
// MI455X (gfx1250) — matrix operand order repaired and checked by reading, not compiled, not run on hardware
//
#include <hip/hip_runtime.h>
#include <math.h>

typedef _Float16 v16h __attribute__((ext_vector_type(16)));
typedef float    v8f  __attribute__((ext_vector_type(8)));

constexpr int DIM   = 256;
constexpr int DEPTH = 5;
constexpr int VOCAB = 257;
constexpr int BB    = 16;
constexpr int TT    = 2048;
constexpr int NEXP  = 8;
constexpr int HID   = 512;
constexpr int TOPK  = 2;
constexpr int CAP   = 5;     // ceil(16*2/8*1.1)
constexpr int HEADS = 2;
constexpr int HD    = DIM / HEADS;

constexpr int KT1 = DIM / 32;     // 8 k-tiles for K=256
constexpr int NT1 = DIM / 16;     // 16 n-tiles for hw1 (N=256)
constexpr int NT2 = 17;           // n-tiles for hw2 (N=257 padded to 272)

constexpr int NSEG = 16;          // context partial-sum segments
constexpr int TSEG = TT / NSEG;   // 128 timesteps per segment

constexpr int LT_STRIDE = 260;    // logit-table row stride (1040 B, 16 B aligned)

// workspace layout (float units)
constexpr int OFF_CTX  = 0;
constexpr int OFF_BASE = OFF_CTX  + BB * DIM;
constexpr int OFF_Y    = OFF_BASE + BB * DIM;
constexpr int OFF_V    = OFF_Y    + BB * DIM;
constexpr int OFF_C    = OFF_V    + BB * DIM;
constexpr int OFF_H    = OFF_C    + BB * DIM;
constexpr int OFF_Z    = OFF_H    + BB * HID;
constexpr int OFF_QKV  = OFF_Z    + BB * DEPTH * DIM;
constexpr int OFF_Z2   = OFF_QKV  + BB * DEPTH * 3 * DIM;
constexpr int OFF_ROOT = OFF_Z2   + BB * DEPTH * DIM;
constexpr int OFF_MOE  = OFF_ROOT + BB * DIM;
constexpr int OFF_PART = OFF_MOE  + BB * DIM;                    // BB*NSEG*DIM
constexpr int OFF_HW1P = OFF_PART + BB * NSEG * DIM;             // f16: NT1*KT1*32*16 halves
constexpr int OFF_HW2P = OFF_HW1P + NT1 * KT1 * 32 * 16 / 2;     // f16: NT2*KT1*32*16 halves
constexpr int OFF_LTAB = OFF_HW2P + NT2 * KT1 * 32 * 16 / 2;     // f32: VOCAB*16*LT_STRIDE

__device__ __forceinline__ float sigmoidf_(float x) { return 1.f / (1.f + expf(-x)); }

// One wave computes a 16x16 f32 tile of A(16xK, f32 row-major) * B(KxN, f32 row-major)
// via v_wmma_f32_16x16x32_f16. Used only for the tiny fractal/MoE kernel where all
// N dims are multiples of 16 (no guards -> no exec-masked loads).
__device__ __forceinline__ v8f wave_tile_mm(const float* __restrict__ A, int lda,
                                            const float* __restrict__ Bw, int ldb,
                                            int K, int n0, v8f c) {
  const int lane = threadIdx.x & 31;
  const int hlf  = lane >> 4;
  const int l16  = lane & 15;
  const int n    = n0 + l16;
  for (int k0 = 0; k0 < K; k0 += 32) {
    v16h a, b;
#pragma unroll
    for (int p = 0; p < 16; ++p) {
      const int ka = k0 + p + 8 * hlf + ((p & 8) ? 8 : 0);  // A frag col (ISA 7.12.2)
      a[p] = (_Float16)A[l16 * lda + ka];
      const int r = k0 + p + 8 * hlf + ((p & 8) ? 8 : 0);   // B frag row
      b[p] = (_Float16)Bw[r * ldb + n];
    }
    c = __builtin_amdgcn_wmma_f32_16x16x32_f16(false, a, false, b, (short)0, c,
                                               false, false);
  }
  return c;
}

// ---------------- kernel 0: pack f32 KxN weight into f16 WMMA B-fragments --------
// Layout: [nt][kt][lane][16 halves]  (32 contiguous bytes per lane per k-step)
__global__ void k_pack_b(const float* __restrict__ Bw, int ldb, int Nvalid,
                         int KT, int NT, _Float16* __restrict__ out) {
  const int idx   = blockIdx.x * blockDim.x + threadIdx.x;
  const int total = NT * KT * 32;
  if (idx >= total) return;
  const int lane = idx & 31;
  const int kt   = (idx >> 5) % KT;
  const int nt   = idx / (32 * KT);
  const int hlf  = lane >> 4;
  const int l16  = lane & 15;
  const int n    = nt * 16 + l16;
  _Float16* dst  = out + (size_t)idx * 16;
  if (n < Nvalid) {
#pragma unroll
    for (int p = 0; p < 16; ++p) {
      const int r = kt * 32 + p + 8 * hlf + ((p & 8) ? 8 : 0);
      dst[p] = (_Float16)Bw[(size_t)r * ldb + n];
    }
  } else {
#pragma unroll
    for (int p = 0; p < 16; ++p) dst[p] = (_Float16)0.f;
  }
}

// ---------------- kernel 1a/1b: context = mean_t embed[tokens] ----------------
// Partial sums over 16 segments (parallel depth 128), then deterministic reduce.
__global__ void k_context_part(const int* __restrict__ tokens,
                               const float* __restrict__ embed,
                               float* __restrict__ part) {
  const int b = blockIdx.x;
  const int s = blockIdx.y;
  const int d = threadIdx.x;
  const int base = b * TT + s * TSEG;
  float acc = 0.f;
  for (int t = 0; t < TSEG; ++t) {
    const int tok = tokens[base + t];
    acc += embed[tok * DIM + d];
  }
  part[(b * NSEG + s) * DIM + d] = acc;
}

__global__ void k_context_red(const float* __restrict__ part,
                              float* __restrict__ ctx) {
  const int b = blockIdx.x;
  const int d = threadIdx.x;
  float acc = 0.f;
#pragma unroll
  for (int s = 0; s < NSEG; ++s) acc += part[(b * NSEG + s) * DIM + d];
  ctx[b * DIM + d] = acc * (1.f / (float)TT);
}

// ---------------- kernel 2: fractal (collapsed) + level attention + MoE ----
__global__ __launch_bounds__(512) void k_fractal_moe(
    float* __restrict__ ws,
    const float* __restrict__ pin_w, const float* __restrict__ pin_b,
    const float* __restrict__ pch_w, const float* __restrict__ pch_b,
    const float* __restrict__ depth_bias,
    const float* __restrict__ v_w, const float* __restrict__ v_b,
    const float* __restrict__ o_w, const float* __restrict__ o_b,
    const float* __restrict__ level_gate, const float* __restrict__ depth_gate,
    const float* __restrict__ in_proj_w, const float* __restrict__ in_proj_b,
    const float* __restrict__ out_proj_w, const float* __restrict__ out_proj_b,
    const float* __restrict__ ln_g, const float* __restrict__ ln_b,
    const float* __restrict__ ic_gate,
    const float* __restrict__ gate_w, const float* __restrict__ gate_b,
    const float* __restrict__ ew1, const float* __restrict__ eb1,
    const float* __restrict__ ew2, const float* __restrict__ eb2) {
  float* ctx   = ws + OFF_CTX;
  float* base  = ws + OFF_BASE;
  float* ybuf  = ws + OFF_Y;
  float* vbuf  = ws + OFF_V;
  float* cbuf  = ws + OFF_C;
  float* hbuf  = ws + OFF_H;
  float* zbuf  = ws + OFF_Z;
  float* qkvb  = ws + OFF_QKV;
  float* z2b   = ws + OFF_Z2;
  float* rootp = ws + OFF_ROOT;
  float* moeb  = ws + OFF_MOE;

  const int tid  = threadIdx.x;
  const int wave = tid >> 5;
  const int lane = tid & 31;
  const int hlf  = lane >> 4;
  const int l16  = lane & 15;

  // base = ctx @ pin_w + pin_b
  {
    v8f c = {};
    c = wave_tile_mm(ctx, DIM, pin_w, DIM, DIM, wave * 16, c);
    const int n = wave * 16 + l16;
#pragma unroll
    for (int j = 0; j < 8; ++j) {
      const int m = j + 8 * hlf;
      base[m * DIM + n] = c[j] + pin_b[n];
    }
  }
  __syncthreads();

  // collapsed fractal: all n-rows per level are identical -> one 16x256 row-set per level
  for (int l = DEPTH - 1; l >= 0; --l) {
    const float gl = sigmoidf_(level_gate[l]);
    if (l == DEPTH - 1) {
      for (int i = tid; i < BB * DIM; i += 512) {
        const int d = i % DIM;
        const float u = (base[i] + depth_bias[l * DIM + d] + pch_b[d]) * gl;
        cbuf[i] = u > 0.f ? u : 0.f;
      }
      __syncthreads();
    } else {
      // vbuf = sigmoid(depth_gate[l]) * (y_next @ v_w + v_b)   (att is exactly uniform)
      {
        v8f c = {};
        c = wave_tile_mm(ybuf, DIM, v_w, DIM, DIM, wave * 16, c);
        const int n = wave * 16 + l16;
        const float dg = sigmoidf_(depth_gate[l * DIM + n]);
#pragma unroll
        for (int j = 0; j < 8; ++j) {
          const int m = j + 8 * hlf;
          vbuf[m * DIM + n] = (c[j] + v_b[n]) * dg;
        }
      }
      __syncthreads();
      // cbuf = relu((base + depth_bias[l] + vbuf @ pch_w + pch_b) * gl)
      {
        v8f c = {};
        c = wave_tile_mm(vbuf, DIM, pch_w, DIM, DIM, wave * 16, c);
        const int n = wave * 16 + l16;
#pragma unroll
        for (int j = 0; j < 8; ++j) {
          const int m = j + 8 * hlf;
          const float u =
              (base[m * DIM + n] + depth_bias[l * DIM + n] + c[j] + pch_b[n]) * gl;
          cbuf[m * DIM + n] = u > 0.f ? u : 0.f;
        }
      }
      __syncthreads();
    }
    // y = relu(upd) @ o_w + o_b ; z[b,l,:] = y
    {
      v8f c = {};
      c = wave_tile_mm(cbuf, DIM, o_w, DIM, DIM, wave * 16, c);
      const int n = wave * 16 + l16;
#pragma unroll
      for (int j = 0; j < 8; ++j) {
        const int m = j + 8 * hlf;
        const float y = c[j] + o_b[n];
        ybuf[m * DIM + n] = y;
        zbuf[(m * DEPTH + l) * DIM + n] = y;
      }
    }
    __syncthreads();
  }

  // qkv = z @ in_proj_w + in_proj_b  (80 rows = 5 M-tiles, 48 N-tiles)
  for (int t = wave; t < 5 * 48; t += 16) {
    const int mt = t / 48, nt = t % 48;
    v8f c = {};
    c = wave_tile_mm(zbuf + mt * 16 * DIM, DIM, in_proj_w, 3 * DIM, DIM, nt * 16, c);
    const int n = nt * 16 + l16;
#pragma unroll
    for (int j = 0; j < 8; ++j) {
      const int m = mt * 16 + j + 8 * hlf;
      qkvb[m * (3 * DIM) + n] = c[j] + in_proj_b[n];
    }
  }
  __syncthreads();

  // attention over DEPTH=5 positions, 2 heads
  __shared__ float s_aw[BB * HEADS * DEPTH * DEPTH];   // 800
  for (int i = tid; i < BB * HEADS * DEPTH * DEPTH; i += 512) {
    const int ki = i % DEPTH;
    const int qi = (i / DEPTH) % DEPTH;
    const int h  = (i / (DEPTH * DEPTH)) % HEADS;
    const int bb = i / (DEPTH * DEPTH * HEADS);
    const float* q = qkvb + (bb * DEPTH + qi) * (3 * DIM) + h * HD;
    const float* k = qkvb + (bb * DEPTH + ki) * (3 * DIM) + DIM + h * HD;
    float acc = 0.f;
    for (int d = 0; d < HD; ++d) acc += q[d] * k[d];
    s_aw[i] = acc * (1.f / sqrtf((float)HD));
  }
  __syncthreads();
  if (tid < BB * HEADS * DEPTH) {
    float* row = s_aw + tid * DEPTH;
    float mx = row[0];
    for (int i = 1; i < DEPTH; ++i) mx = fmaxf(mx, row[i]);
    float s = 0.f;
    for (int i = 0; i < DEPTH; ++i) { row[i] = expf(row[i] - mx); s += row[i]; }
    const float inv = 1.f / s;
    for (int i = 0; i < DEPTH; ++i) row[i] *= inv;
  }
  __syncthreads();
  for (int i = tid; i < BB * DEPTH * DIM; i += 512) {
    const int d  = i % DIM;
    const int qi = (i / DIM) % DEPTH;
    const int bb = i / (DIM * DEPTH);
    const int h  = d / HD;
    const int d2 = d % HD;
    const float* aw = s_aw + ((bb * HEADS + h) * DEPTH + qi) * DEPTH;
    float acc = 0.f;
    for (int ki = 0; ki < DEPTH; ++ki)
      acc += aw[ki] * qkvb[(bb * DEPTH + ki) * (3 * DIM) + 2 * DIM + h * HD + d2];
    z2b[i] = acc;
  }
  __syncthreads();

  // z2 = z2 @ out_proj_w + out_proj_b  (qkv buffer reused as destination)
  float* zproj = qkvb;
  for (int t = wave; t < 5 * 16; t += 16) {
    const int mt = t / 16, nt = t % 16;
    v8f c = {};
    c = wave_tile_mm(z2b + mt * 16 * DIM, DIM, out_proj_w, DIM, DIM, nt * 16, c);
    const int n = nt * 16 + l16;
#pragma unroll
    for (int j = 0; j < 8; ++j) {
      const int m = mt * 16 + j + 8 * hlf;
      zproj[m * DIM + n] = c[j] + out_proj_b[n];
    }
  }
  __syncthreads();

  // layernorm on row l=0 of each batch; root = y0 + sigmoid(ic_gate)*zn
  if (tid < BB) {
    const int bb = tid;
    const int r  = bb * DEPTH;   // l = 0
    float mu = 0.f;
    for (int d = 0; d < DIM; ++d) mu += zbuf[r * DIM + d] + zproj[r * DIM + d];
    mu *= (1.f / (float)DIM);
    float var = 0.f;
    for (int d = 0; d < DIM; ++d) {
      const float t2 = zbuf[r * DIM + d] + zproj[r * DIM + d] - mu;
      var += t2 * t2;
    }
    var *= (1.f / (float)DIM);
    const float inv = rsqrtf(var + 1e-5f);
    const float icg = sigmoidf_(ic_gate[0]);
    for (int d = 0; d < DIM; ++d) {
      const float zs = zbuf[r * DIM + d] + zproj[r * DIM + d];
      const float zn = (zs - mu) * inv * ln_g[d] + ln_b[d];
      rootp[bb * DIM + d] = zbuf[r * DIM + d] + icg * zn;
    }
  }
  __syncthreads();

  // ---------------- MoE ----------------
  __shared__ float s_sc[BB * NEXP];
  __shared__ float s_gs[BB * TOPK];
  __shared__ int   s_gi[BB * TOPK];
  __shared__ float s_w[NEXP * CAP];
  __shared__ int   s_tok[NEXP * CAP];

  if (tid < BB * NEXP) {
    const int bb = tid / NEXP, e = tid % NEXP;
    float acc = gate_b[e];
    for (int d = 0; d < DIM; ++d) acc += rootp[bb * DIM + d] * gate_w[d * NEXP + e];
    s_sc[tid] = acc;   // TEMP = 1
  }
  __syncthreads();
  if (tid < BB) {
    const float* sc = s_sc + tid * NEXP;
    int i1 = 0; float v1 = sc[0];
    for (int e = 1; e < NEXP; ++e) if (sc[e] > v1) { v1 = sc[e]; i1 = e; }
    int i2 = (i1 == 0) ? 1 : 0; float v2 = sc[i2];
    for (int e = 0; e < NEXP; ++e)
      if (e != i1 && sc[e] > v2) { v2 = sc[e]; i2 = e; }
    const float e2 = expf(v2 - v1);
    s_gs[tid * 2 + 0] = 1.f / (1.f + e2);
    s_gs[tid * 2 + 1] = e2 / (1.f + e2);
    s_gi[tid * 2 + 0] = i1;
    s_gi[tid * 2 + 1] = i2;
  }
  __syncthreads();
  if (tid < NEXP) {
    const int e = tid;
    unsigned used = 0u;
    for (int c2 = 0; c2 < CAP; ++c2) {
      float best = -1e30f; int bi = 0;
      for (int i = 0; i < BB * TOPK; ++i) {
        if (used & (1u << i)) continue;
        const float val = (s_gi[i] == e) ? s_gs[i] : -1.f;
        if (val > best) { best = val; bi = i; }
      }
      used |= (1u << bi);
      s_w[e * CAP + c2]   = best > 0.f ? best : 0.f;
      s_tok[e * CAP + c2] = bi / TOPK;
    }
  }
  __syncthreads();

  for (int i = tid; i < BB * DIM; i += 512) moeb[i] = 0.f;
  __syncthreads();

  for (int e = 0; e < NEXP; ++e) {
    // xin (CAP rows, zero-padded to 16) in cbuf
    for (int i = tid; i < 16 * DIM; i += 512) {
      const int row = i / DIM, d = i % DIM;
      cbuf[i] = (row < CAP) ? rootp[s_tok[e * CAP + row] * DIM + d] : 0.f;
    }
    __syncthreads();
    const float* w1 = ew1 + (size_t)e * DIM * HID;
    const float* b1 = eb1 + e * HID;
    for (int t = wave; t < HID / 16; t += 16) {      // 32 N-tiles
      v8f c = {};
      c = wave_tile_mm(cbuf, DIM, w1, HID, DIM, t * 16, c);
      const int n = t * 16 + l16;
#pragma unroll
      for (int j = 0; j < 8; ++j) {
        const int m = j + 8 * hlf;
        const float u = c[j] + b1[n];
        hbuf[m * HID + n] = u > 0.f ? u : 0.f;
      }
    }
    __syncthreads();
    const float* w2 = ew2 + (size_t)e * HID * DIM;
    const float* b2 = eb2 + e * DIM;
    {
      v8f c = {};
      c = wave_tile_mm(hbuf, HID, w2, DIM, HID, wave * 16, c);
      const int n = wave * 16 + l16;
#pragma unroll
      for (int j = 0; j < 8; ++j) {
        const int m = j + 8 * hlf;
        vbuf[m * DIM + n] = c[j] + b2[n];
      }
    }
    __syncthreads();
    for (int i = tid; i < CAP * DIM; i += 512) {
      const int row = i / DIM, d = i % DIM;
      const float wv = s_w[e * CAP + row];
      if (wv != 0.f)
        moeb[s_tok[e * CAP + row] * DIM + d] += vbuf[row * DIM + d] * wv;
    }
    __syncthreads();
  }
  for (int i = tid; i < BB * DIM; i += 512) rootp[i] += moeb[i];
}

// ------------- kernel 3: per-vocab fused table GEMM (the big compute) -------
// A/U staged in LDS as pre-packed f16 fragments; B read as packed fragments
// (2x ds_load_b128 + 2x global_load_b128 + 1x v_wmma per k-step).
__global__ __launch_bounds__(512) void k_table(
    const float* __restrict__ embed, const float* __restrict__ rootp,
    const _Float16* __restrict__ hw1p, const float* __restrict__ hb1,
    const _Float16* __restrict__ hw2p, const float* __restrict__ hb2,
    float* __restrict__ Ltab) {
  __shared__ float sF[16 * DIM];                         // f32 staging (A, then U)
  __shared__ __align__(32) _Float16 sAf[KT1 * 32 * 16];  // packed A fragments (8 KB)
  __shared__ __align__(32) _Float16 sUf[KT1 * 32 * 16];  // packed U fragments (8 KB)
  const int v    = blockIdx.x;
  const int tid  = threadIdx.x;
  const int wave = tid >> 5;
  const int lane = tid & 31;
  const int hlf  = lane >> 4;
  const int l16  = lane & 15;

  // stage A = embed[v] + root (16 x 256)
  for (int i = tid; i < 16 * DIM; i += 512) {
    const int bb = i / DIM, k = i % DIM;
    sF[i] = embed[v * DIM + k] + rootp[bb * DIM + k];
  }
  __syncthreads();
  // pack A into f16 fragments: [kt][lane][p], elem p -> col kt*32 + p + 8*hlf + (p&8?8:0)
  for (int idx = tid; idx < KT1 * 32 * 16; idx += 512) {
    const int p  = idx & 15;
    const int ln = (idx >> 4) & 31;
    const int kt = idx >> 9;
    const int h2 = ln >> 4, r2 = ln & 15;
    const int k  = kt * 32 + p + 8 * h2 + ((p & 8) ? 8 : 0);
    sAf[idx] = (_Float16)sF[r2 * DIM + k];
  }
  __syncthreads();

  // GEMM1: U = relu(A @ hw1 + hb1); one n-tile per wave
  {
    const v16h* ap = (const v16h*)sAf;
    const v16h* bp = (const v16h*)hw1p + ((size_t)wave * KT1) * 32 + lane;
    v8f c = {};
#pragma unroll
    for (int kt = 0; kt < KT1; ++kt)
      c = __builtin_amdgcn_wmma_f32_16x16x32_f16(false, ap[kt * 32 + lane], false,
                                                 bp[kt * 32], (short)0, c, false,
                                                 false);
    const int n = wave * 16 + l16;
#pragma unroll
    for (int j = 0; j < 8; ++j) {
      const int m = j + 8 * hlf;
      const float u = c[j] + hb1[n];
      sF[m * DIM + n] = u > 0.f ? u : 0.f;
    }
  }
  __syncthreads();
  // pack U fragments
  for (int idx = tid; idx < KT1 * 32 * 16; idx += 512) {
    const int p  = idx & 15;
    const int ln = (idx >> 4) & 31;
    const int kt = idx >> 9;
    const int h2 = ln >> 4, r2 = ln & 15;
    const int k  = kt * 32 + p + 8 * h2 + ((p & 8) ? 8 : 0);
    sUf[idx] = (_Float16)sF[r2 * DIM + k];
  }
  __syncthreads();

  // GEMM2: L = U @ hw2 + hb2  (17 n-tiles, weights pre-padded -> unguarded loads)
  for (int t = wave; t < NT2; t += 16) {
    const v16h* ap = (const v16h*)sUf;
    const v16h* bp = (const v16h*)hw2p + ((size_t)t * KT1) * 32 + lane;
    v8f c = {};
#pragma unroll
    for (int kt = 0; kt < KT1; ++kt)
      c = __builtin_amdgcn_wmma_f32_16x16x32_f16(false, ap[kt * 32 + lane], false,
                                                 bp[kt * 32], (short)0, c, false,
                                                 false);
    const int n = t * 16 + l16;
    if (n < VOCAB) {
#pragma unroll
      for (int j = 0; j < 8; ++j) {
        const int m = j + 8 * hlf;
        Ltab[((size_t)v * 16 + m) * LT_STRIDE + n] = c[j] + hb2[n];
      }
    }
  }
}

// ------------- kernel 4: gather table rows into (B,T,VOCAB) output ---------
// Table rows are 16 B aligned (stride 260 floats): 64 aligned b128 loads + tail,
// scalar b32 stores (output rows are stride-1028 B, only 4 B aligned).
__global__ __launch_bounds__(64) void k_gather(const int* __restrict__ tokens,
                                               const float* __restrict__ Ltab,
                                               float* __restrict__ out) {
  const int idx = blockIdx.x;           // b*T + t
  const int b   = idx >> 11;            // T = 2048
  const int tok = tokens[idx];
  const float* srcf = Ltab + ((size_t)tok * 16 + b) * LT_STRIDE;
  const float4* src4 = (const float4*)srcf;
  float* dst = out + (size_t)idx * VOCAB;
  const int t = threadIdx.x;            // 0..63
  const float4 v4 = src4[t];
  dst[4 * t + 0] = v4.x;
  dst[4 * t + 1] = v4.y;
  dst[4 * t + 2] = v4.z;
  dst[4 * t + 3] = v4.w;
  if (t == 0) dst[256] = srcf[256];
}

extern "C" void kernel_launch(void* const* d_in, const int* in_sizes, int n_in,
                              void* d_out, int out_size, void* d_ws, size_t ws_size,
                              hipStream_t stream) {
  const int*   tokens     = (const int*)  d_in[0];
  const float* embed      = (const float*)d_in[1];
  const float* pin_w      = (const float*)d_in[2];
  const float* pin_b      = (const float*)d_in[3];
  const float* pch_w      = (const float*)d_in[4];
  const float* pch_b      = (const float*)d_in[5];
  const float* depth_bias = (const float*)d_in[6];
  // d_in[7..10] = q_w, q_b, k_w, k_b: mathematically eliminated (uniform softmax)
  const float* v_w        = (const float*)d_in[11];
  const float* v_b        = (const float*)d_in[12];
  const float* o_w        = (const float*)d_in[13];
  const float* o_b        = (const float*)d_in[14];
  const float* level_gate = (const float*)d_in[15];
  const float* depth_gate = (const float*)d_in[16];
  const float* in_proj_w  = (const float*)d_in[17];
  const float* in_proj_b  = (const float*)d_in[18];
  const float* out_proj_w = (const float*)d_in[19];
  const float* out_proj_b = (const float*)d_in[20];
  const float* ln_g       = (const float*)d_in[21];
  const float* ln_b       = (const float*)d_in[22];
  const float* ic_gate    = (const float*)d_in[23];
  const float* gate_w     = (const float*)d_in[24];
  const float* gate_b     = (const float*)d_in[25];
  const float* ew1        = (const float*)d_in[26];
  const float* eb1        = (const float*)d_in[27];
  const float* ew2        = (const float*)d_in[28];
  const float* eb2        = (const float*)d_in[29];
  const float* hw1        = (const float*)d_in[30];
  const float* hb1        = (const float*)d_in[31];
  const float* hw2        = (const float*)d_in[32];
  const float* hb2        = (const float*)d_in[33];

  float*     ws    = (float*)d_ws;
  float*     ctx   = ws + OFF_CTX;
  float*     part  = ws + OFF_PART;
  float*     rootp = ws + OFF_ROOT;
  _Float16*  hw1p  = (_Float16*)(ws + OFF_HW1P);
  _Float16*  hw2p  = (_Float16*)(ws + OFF_HW2P);
  float*     Ltab  = ws + OFF_LTAB;

  // pack final-head weights into WMMA fragment layout (one pass, reused by 257 blocks)
  {
    const int tot1 = NT1 * KT1 * 32;
    k_pack_b<<<(tot1 + 255) / 256, 256, 0, stream>>>(hw1, DIM, DIM, KT1, NT1, hw1p);
    const int tot2 = NT2 * KT1 * 32;
    k_pack_b<<<(tot2 + 255) / 256, 256, 0, stream>>>(hw2, VOCAB, VOCAB, KT1, NT2, hw2p);
  }
  k_context_part<<<dim3(BB, NSEG), DIM, 0, stream>>>(tokens, embed, part);
  k_context_red<<<BB, DIM, 0, stream>>>(part, ctx);
  k_fractal_moe<<<1, 512, 0, stream>>>(ws, pin_w, pin_b, pch_w, pch_b, depth_bias,
                                       v_w, v_b, o_w, o_b, level_gate, depth_gate,
                                       in_proj_w, in_proj_b, out_proj_w, out_proj_b,
                                       ln_g, ln_b, ic_gate, gate_w, gate_b,
                                       ew1, eb1, ew2, eb2);
  k_table<<<VOCAB, 512, 0, stream>>>(embed, rootp, hw1p, hb1, hw2p, hb2, Ltab);
  k_gather<<<BB * TT, 64, 0, stream>>>(tokens, Ltab, (float*)d_out);
}
